// Attention_54992761258239
// MI455X (gfx1250) — matrix operand order repaired and checked by reading, compiled, not run on hardware
//
#include <hip/hip_runtime.h>
#include <hip/hip_bf16.h>

typedef __attribute__((ext_vector_type(16))) __bf16 v16bf;
typedef __attribute__((ext_vector_type(8)))  __bf16 v8bf;
typedef __attribute__((ext_vector_type(8)))  float  v8f;

#define DIMC   768
#define NH     16
#define HD     48
#define DPAD   64
#define BATCH  4
#define SEQ    1568
#define ROWS   (BATCH*SEQ)      /* 6272 */
#define QKVC   (3*DIMC)         /* 2304 */

__device__ __forceinline__ v16bf cat8(v8bf lo, v8bf hi) {
  return __builtin_shufflevector(lo, hi, 0,1,2,3,4,5,6,7,8,9,10,11,12,13,14,15);
}

__device__ __forceinline__ v8f wmma_zero() {
  v8f z = {0.f,0.f,0.f,0.f,0.f,0.f,0.f,0.f};
  return z;
}

// One 16x16 f32 tile of A(16xK) * Bt(16xK)^T accumulated over K in steps of 32
// using v_wmma_f32_16x16x32_bf16.  A-operand: lane = row (lane%16), K split by
// half-wave in 16-elem runs.  B-operand: lane = column, K split by half-wave in
// 16-elem runs (Bt is stored column-major, i.e. Bt row = output column).
__device__ __forceinline__ v8f wmma_row_tile(const __bf16* __restrict__ Arow,
                                             const __bf16* __restrict__ Btrow,
                                             int Kdim, int hf) {
  v8f acc = wmma_zero();
  for (int k0 = 0; k0 < Kdim; k0 += 32) {
    __builtin_prefetch((const void*)(Arow + k0 + 256), 0, 1);
    __builtin_prefetch((const void*)(Btrow + k0 + 256), 0, 1);
    v8bf a0 = *(const v8bf*)(Arow + k0 + hf*16);
    v8bf a1 = *(const v8bf*)(Arow + k0 + hf*16 + 8);
    v8bf b0 = *(const v8bf*)(Btrow + k0 + hf*16);
    v8bf b1 = *(const v8bf*)(Btrow + k0 + hf*16 + 8);
    acc = __builtin_amdgcn_wmma_f32_16x16x32_bf16(false, cat8(a0,a1),
                                                  false, cat8(b0,b1),
                                                  (short)0, acc, false, false);
  }
  return acc;
}

// ---------------- conversion / init kernels ----------------

__global__ __launch_bounds__(256) void k_cvt_bf16(const float* __restrict__ src,
                                                  __bf16* __restrict__ dst, int n) {
  int i = blockIdx.x*256 + threadIdx.x;
  if (i < n) dst[i] = (__bf16)src[i];
}

// src: R x C fp32, dst: C x R bf16  (dst[c][k] = src[k][c])
__global__ __launch_bounds__(256) void k_cvt_transpose(const float* __restrict__ src,
                                                       __bf16* __restrict__ dst,
                                                       int R, int C) {
  int i = blockIdx.x*256 + threadIdx.x;
  if (i < R*C) {
    int c = i / R;
    int k = i % R;
    dst[(size_t)c*R + k] = (__bf16)src[(size_t)k*C + c];
  }
}

__global__ __launch_bounds__(256) void k_zero4(uint4* __restrict__ p, int n4) {
  int i = blockIdx.x*256 + threadIdx.x;
  if (i < n4) p[i] = make_uint4(0u,0u,0u,0u);
}

// ---------------- QKV projection GEMM ----------------
// X(6272x768)bf16 @ Wqkv_t(2304x768)bf16^T
//   -> Q,K scattered key/query-major [B,H,N,64]  (A / B-as-rows operands)
//   -> V scattered dim-major        [B,H,64,N]  (direct PV B-operand stream)

__global__ __launch_bounds__(128) void k_gemm_qkv(const __bf16* __restrict__ X,
                                                  const __bf16* __restrict__ Wt,
                                                  __bf16* __restrict__ Q,
                                                  __bf16* __restrict__ Kb,
                                                  __bf16* __restrict__ Vt) {
  int wave = threadIdx.x >> 5, lane = threadIdx.x & 31;
  int hf = lane >> 4, l16 = lane & 15;
  int gw = blockIdx.x*4 + wave;               // 0..56447
  int tm = gw / (QKVC/16);                    // / 144
  int tn = gw % (QKVC/16);
  const __bf16* Arow = X  + (size_t)(tm*16 + l16)*DIMC;
  const __bf16* Brow = Wt + (size_t)(tn*16 + l16)*DIMC;
  v8f acc = wmma_row_tile(Arow, Brow, DIMC, hf);

  int cc   = tn*16 + l16;
  int sec  = cc / DIMC;
  int rem  = cc % DIMC;
  int head = rem / HD;
  int d    = rem % HD;
#pragma unroll
  for (int v = 0; v < 8; ++v) {
    int r = tm*16 + v + 8*hf;                 // output row
    int b = r / SEQ, n = r % SEQ;
    __bf16 val = (__bf16)acc[v];
    if (sec == 0) {
      Q [(((size_t)b*NH + head)*SEQ + n)*DPAD + d] = val;
    } else if (sec == 1) {
      Kb[(((size_t)b*NH + head)*SEQ + n)*DPAD + d] = val;
    } else {
      Vt[(((size_t)b*NH + head)*DPAD + d)*SEQ + n] = val;   // transposed
    }
  }
}

// ---------------- RoPE (3-axis) on Q and K ----------------
// The attention score scale 1/sqrt(48) is folded into Q here (its pad region
// stays zero, and all 48 live dims pass through this kernel).

__global__ __launch_bounds__(256) void k_rope(__bf16* __restrict__ Q,
                                              __bf16* __restrict__ Kb,
                                              const int* __restrict__ pos) {
  int t = blockIdx.x*256 + threadIdx.x;       // total = 2*4*16*1568*3*8
  int j = t & 7;  t >>= 3;
  int axis = t % 3;  t /= 3;
  int n = t % SEQ;  t /= SEQ;
  int h = t % NH;   t /= NH;
  int b = t % BATCH; t /= BATCH;
  int isK = t;                                // 0 = Q, 1 = K
  __bf16* buf = (isK == 0) ? Q : Kb;
  float scl = (isK == 0) ? 0.14433756729740643f : 1.0f;   // 1/sqrt(48) on Q
  float p = (float)pos[n*3 + axis];
  // inv_freq = 10000^(-j/8) = exp(-j * ln(10000)/8)
  float freq = __expf(-(float)j * (9.210340371976184f * 0.125f));
  float ang = p * freq;
  float s, c;
  __sincosf(ang, &s, &c);
  size_t base = (((size_t)b*NH + h)*SEQ + n)*DPAD + axis*16;
  float t1 = (float)buf[base + j];
  float t2 = (float)buf[base + 8 + j];
  buf[base + j]     = (__bf16)((t1*c - t2*s)*scl);
  buf[base + 8 + j] = (__bf16)((t2*c + t1*s)*scl);
}

// ---------------- Flash attention ----------------
// One wave handles a 16-query tile of one (b,h); 32 keys per step.
// K is key-major (B-operand rows = key rows); V is dim-major so the PV
// B-operand is a contiguous 16-half run per lane (two b128 loads).
// Row sums of P are computed by a WMMA against a ones matrix (the D layout
// drops rowsum(m) into the same per-lane slots as the lst[] state), keeping
// the softmax bookkeeping off the VALU shuffle path.

__global__ __launch_bounds__(128) void k_attn(const __bf16* __restrict__ Q,
                                              const __bf16* __restrict__ K,
                                              const __bf16* __restrict__ Vt,
                                              __bf16* __restrict__ O) {
  __shared__ __align__(16) __bf16 ldsP[4][16*32];

  int wave = threadIdx.x >> 5, lane = threadIdx.x & 31;
  int hf = lane >> 4, l16 = lane & 15;
  int gw = blockIdx.x*4 + wave;               // 0..6271
  int qt = gw % (SEQ/16);                     // 98 query tiles
  int bh = gw / (SEQ/16);                     // 0..63
  const __bf16* Qb  = Q  + (size_t)bh * SEQ * DPAD;
  const __bf16* Kbb = K  + (size_t)bh * SEQ * DPAD;
  const __bf16* Vb  = Vt + (size_t)bh * SEQ * DPAD;  // [64][1568] dim-major
  __bf16* myP = ldsP[wave];

  // constant ones B-operand for row-sum WMMA
  v16bf ones;
#pragma unroll
  for (int i = 0; i < 16; ++i) ones[i] = (__bf16)1.0f;

  // Q tile as two A-operands (K = 0..31, 32..63)
  const __bf16* qp = Qb + (size_t)(qt*16 + l16)*DPAD;
  v16bf qa0 = cat8(*(const v8bf*)(qp + hf*16),      *(const v8bf*)(qp + hf*16 + 8));
  v16bf qa1 = cat8(*(const v8bf*)(qp + 32 + hf*16), *(const v8bf*)(qp + 32 + hf*16 + 8));

  // Per-lane base pointers for the 4 Vt rows this lane streams (d = dt*16+l16)
  const __bf16* vrow0 = Vb + (size_t)(0*16 + l16)*SEQ;
  const __bf16* vrow1 = Vb + (size_t)(1*16 + l16)*SEQ;
  const __bf16* vrow2 = Vb + (size_t)(2*16 + l16)*SEQ;
  const __bf16* vrow3 = Vb + (size_t)(3*16 + l16)*SEQ;

  v8f oacc[4];
  oacc[0] = wmma_zero(); oacc[1] = wmma_zero();
  oacc[2] = wmma_zero(); oacc[3] = wmma_zero();
  float mst[8], lst[8], corr[8];
#pragma unroll
  for (int v = 0; v < 8; ++v) { mst[v] = -3.0e38f; lst[v] = 0.f; }

  for (int kb = 0; kb < SEQ; kb += 32) {
    // ---- S = Q K^T for two 16-key tiles (4 WMMAs); scale pre-folded in Q ----
    const __bf16* kp0 = Kbb + (size_t)(kb + l16)*DPAD;
    const __bf16* kp1 = kp0 + (size_t)16*DPAD;
    __builtin_prefetch((const void*)(kp0 + 32*DPAD), 0, 1);
    v16bf kt00 = cat8(*(const v8bf*)(kp0 + hf*16),      *(const v8bf*)(kp0 + hf*16 + 8));
    v16bf kt01 = cat8(*(const v8bf*)(kp0 + 32 + hf*16), *(const v8bf*)(kp0 + 32 + hf*16 + 8));
    v16bf kt10 = cat8(*(const v8bf*)(kp1 + hf*16),      *(const v8bf*)(kp1 + hf*16 + 8));
    v16bf kt11 = cat8(*(const v8bf*)(kp1 + 32 + hf*16), *(const v8bf*)(kp1 + 32 + hf*16 + 8));
    v8f s0 = wmma_zero();
    s0 = __builtin_amdgcn_wmma_f32_16x16x32_bf16(false, qa0, false, kt00, (short)0, s0, false, false);
    s0 = __builtin_amdgcn_wmma_f32_16x16x32_bf16(false, qa1, false, kt01, (short)0, s0, false, false);
    v8f s1 = wmma_zero();
    s1 = __builtin_amdgcn_wmma_f32_16x16x32_bf16(false, qa0, false, kt10, (short)0, s1, false, false);
    s1 = __builtin_amdgcn_wmma_f32_16x16x32_bf16(false, qa1, false, kt11, (short)0, s1, false, false);

    // ---- prefetch next Vt runs while softmax runs ----
    __builtin_prefetch((const void*)(vrow0 + kb + 32), 0, 1);
    __builtin_prefetch((const void*)(vrow3 + kb + 32), 0, 1);

    // ---- online softmax max/exp (row = v + 8*hf, col = l16) ----
#pragma unroll
    for (int v = 0; v < 8; ++v) {
      float a = s0[v], b = s1[v];
      float mx = fmaxf(a, b);
      for (int off = 1; off < 16; off <<= 1)
        mx = fmaxf(mx, __shfl_xor(mx, off, 32));
      float mnew = fmaxf(mst[v], mx);
      corr[v] = __expf(mst[v] - mnew);
      mst[v] = mnew;
      float p0 = __expf(a - mnew);
      float p1 = __expf(b - mnew);
      oacc[0][v] *= corr[v]; oacc[1][v] *= corr[v];
      oacc[2][v] *= corr[v]; oacc[3][v] *= corr[v];
      int row = v + 8*hf;
      myP[row*32 + l16]      = (__bf16)p0;
      myP[row*32 + 16 + l16] = (__bf16)p1;
    }

    // ---- P as A-operand (16x32, via LDS re-layout) ----
    v16bf pa = cat8(*(const v8bf*)(myP + l16*32 + hf*16),
                    *(const v8bf*)(myP + l16*32 + hf*16 + 8));

    // ---- row sums of P via WMMA against ones (replaces shuffle reduction) ----
    v8f rs = __builtin_amdgcn_wmma_f32_16x16x32_bf16(false, pa, false, ones,
                                                     (short)0, wmma_zero(), false, false);
#pragma unroll
    for (int v = 0; v < 8; ++v)
      lst[v] = lst[v]*corr[v] + rs[v];

    // ---- O += P V : B-operand streamed from dim-major Vt (4 WMMAs) ----
    v16bf vb0 = cat8(*(const v8bf*)(vrow0 + kb + hf*16), *(const v8bf*)(vrow0 + kb + hf*16 + 8));
    v16bf vb1 = cat8(*(const v8bf*)(vrow1 + kb + hf*16), *(const v8bf*)(vrow1 + kb + hf*16 + 8));
    v16bf vb2 = cat8(*(const v8bf*)(vrow2 + kb + hf*16), *(const v8bf*)(vrow2 + kb + hf*16 + 8));
    v16bf vb3 = cat8(*(const v8bf*)(vrow3 + kb + hf*16), *(const v8bf*)(vrow3 + kb + hf*16 + 8));
    oacc[0] = __builtin_amdgcn_wmma_f32_16x16x32_bf16(false, pa, false, vb0, (short)0, oacc[0], false, false);
    oacc[1] = __builtin_amdgcn_wmma_f32_16x16x32_bf16(false, pa, false, vb1, (short)0, oacc[1], false, false);
    oacc[2] = __builtin_amdgcn_wmma_f32_16x16x32_bf16(false, pa, false, vb2, (short)0, oacc[2], false, false);
    oacc[3] = __builtin_amdgcn_wmma_f32_16x16x32_bf16(false, pa, false, vb3, (short)0, oacc[3], false, false);
  }

  // ---- finalize and write [B,N,768] bf16 (skip dim pad 48..63) ----
  int b = bh >> 4, h = bh & 15;
#pragma unroll
  for (int v = 0; v < 8; ++v) {
    float inv = 1.0f / lst[v];
    int qrow = qt*16 + v + 8*hf;
#pragma unroll
    for (int dt = 0; dt < 3; ++dt) {
      int d = dt*16 + l16;
      O[((size_t)b*SEQ + qrow)*DIMC + h*HD + d] = (__bf16)(oacc[dt][v]*inv);
    }
  }
}

// ---------------- output projection GEMM + bias ----------------

__global__ __launch_bounds__(128) void k_gemm_proj(const __bf16* __restrict__ X,
                                                   const __bf16* __restrict__ Wt,
                                                   const float* __restrict__ bias,
                                                   float* __restrict__ out) {
  int wave = threadIdx.x >> 5, lane = threadIdx.x & 31;
  int hf = lane >> 4, l16 = lane & 15;
  int gw = blockIdx.x*4 + wave;               // tiles: 392*48 = 18816
  int tm = gw / (DIMC/16);
  int tn = gw % (DIMC/16);
  v8f acc = wmma_row_tile(X  + (size_t)(tm*16 + l16)*DIMC,
                          Wt + (size_t)(tn*16 + l16)*DIMC, DIMC, hf);
  int c = tn*16 + l16;
  float bb = bias[c];
#pragma unroll
  for (int v = 0; v < 8; ++v) {
    int r = tm*16 + v + 8*hf;
    out[(size_t)r*DIMC + c] = acc[v] + bb;
  }
}

// ---------------- host launcher ----------------

extern "C" void kernel_launch(void* const* d_in, const int* in_sizes, int n_in,
                              void* d_out, int out_size, void* d_ws, size_t ws_size,
                              hipStream_t stream) {
  (void)in_sizes; (void)n_in; (void)out_size; (void)ws_size;
  const float* x     = (const float*)d_in[0];
  const int*   pos   = (const int*)d_in[1];
  const float* Wqkv  = (const float*)d_in[2];
  const float* Wproj = (const float*)d_in[3];
  const float* bproj = (const float*)d_in[4];
  float* out = (float*)d_out;

  char* ws = (char*)d_ws;
  size_t off = 0;
  auto take = [&](size_t bytes) -> char* {
    char* p = ws + off;
    off += (bytes + 255) & ~(size_t)255;
    return p;
  };

  __bf16* Xb      = (__bf16*)take((size_t)ROWS*DIMC*2);
  __bf16* Wqkv_t  = (__bf16*)take((size_t)QKVC*DIMC*2);
  __bf16* Wproj_t = (__bf16*)take((size_t)DIMC*DIMC*2);
  size_t qkvElems = (size_t)BATCH*NH*SEQ*DPAD;          // 6,422,528 per tensor
  __bf16* Qb = (__bf16*)take(qkvElems*2*3);             // Q,K,Vt contiguous
  __bf16* Kb = Qb + qkvElems;
  __bf16* Vt = Kb + qkvElems;
  __bf16* Ob = (__bf16*)take((size_t)ROWS*DIMC*2);

  // 1) conversions to bf16 (weights pre-transposed to column-major)
  k_cvt_bf16<<<(ROWS*DIMC + 255)/256, 256, 0, stream>>>(x, Xb, ROWS*DIMC);
  k_cvt_transpose<<<(DIMC*QKVC + 255)/256, 256, 0, stream>>>(Wqkv, Wqkv_t, DIMC, QKVC);
  k_cvt_transpose<<<(DIMC*DIMC + 255)/256, 256, 0, stream>>>(Wproj, Wproj_t, DIMC, DIMC);

  // 2) zero Q/K/Vt (head-dim padded 48 -> 64)
  int z4 = (int)(qkvElems*3*2/16);                      // uint4 count (exact)
  k_zero4<<<(z4 + 255)/256, 256, 0, stream>>>((uint4*)Qb, z4);

  // 3) QKV projection (WMMA); V written transposed (dim-major)
  k_gemm_qkv<<<(ROWS/16)*(QKVC/16)/4, 128, 0, stream>>>(Xb, Wqkv_t, Qb, Kb, Vt);

  // 4) RoPE on Q, K (+ fold 1/sqrt(48) into Q)
  k_rope<<<(2*BATCH*NH*SEQ*3*8)/256, 256, 0, stream>>>(Qb, Kb, pos);

  // 5) flash attention (WMMA for QK^T, PV, and softmax row sums)
  k_attn<<<(BATCH*NH*(SEQ/16))/4, 128, 0, stream>>>(Qb, Kb, Vt, Ob);

  // 6) output projection + bias (WMMA), f32 output
  k_gemm_proj<<<(ROWS/16)*(DIMC/16)/4, 128, 0, stream>>>(Ob, Wproj_t, bproj, out);
}
